// MoE_dist_48653389529292
// MI455X (gfx1250) — matrix operand order repaired and checked by reading, not compiled, not run on hardware
//
#include <hip/hip_runtime.h>
#include <hip/hip_bf16.h>

// ---------------- problem constants (match reference setup_inputs) ----------
#define BDIM 8
#define TDIM 2048
#define CDIM 1024
#define HDIM 4096
#define EEXP 8
#define KTOP 2
#define CAP  320              // int(T/E * 1.25)
#define NROW (BDIM*CAP)       // 2560 padded rows per expert
#define TKN  (TDIM*KTOP)      // assignments per batch row

#if __has_builtin(__builtin_amdgcn_tensor_load_to_lds)
#define HAVE_TDM 1
#else
#define HAVE_TDM 0
#endif

// ---------------- vector types ----------------------------------------------
typedef __attribute__((ext_vector_type(16))) __bf16 bf16x16;
typedef __attribute__((ext_vector_type(8)))  __bf16 bf16x8;
typedef __attribute__((ext_vector_type(4)))  __bf16 bf16x4;
typedef __attribute__((ext_vector_type(8)))  float  f32x8;
typedef __attribute__((ext_vector_type(4)))  float  f32x4v;
typedef __attribute__((ext_vector_type(4)))  unsigned int u32x4;
typedef __attribute__((ext_vector_type(8)))  int    i32x8;
typedef __attribute__((ext_vector_type(4)))  int    i32x4;

// ---------------- fp32 -> bf16 weight conversion (128-bit traffic) ----------
__global__ void cvt_f32_bf16(const float* __restrict__ in, __bf16* __restrict__ out,
                             size_t n4) {                 // n4 = n/4
  size_t i = (size_t)blockIdx.x * blockDim.x + threadIdx.x;
  size_t stride = (size_t)gridDim.x * blockDim.x;
  for (; i < n4; i += stride) {
    f32x4v v = ((const f32x4v*)in)[i];
    bf16x4 o;
#pragma unroll
    for (int j = 0; j < 4; ++j) o[j] = (__bf16)v[j];
    ((bf16x4*)out)[i] = o;
  }
}

__global__ void zero_b128(uint4* __restrict__ p, size_t n16) {
  size_t i = (size_t)blockIdx.x * blockDim.x + threadIdx.x;
  size_t stride = (size_t)gridDim.x * blockDim.x;
  uint4 z; z.x = z.y = z.z = z.w = 0u;
  for (; i < n16; i += stride) p[i] = z;
}

// ---------------- router: logits -> softmax -> top2 (one wave per token) ----
__global__ __launch_bounds__(256) void router_topk(
    const float* __restrict__ x, const float* __restrict__ Wr,
    const float* __restrict__ br, int* __restrict__ tke, float* __restrict__ tkp) {
  __shared__ float sW[CDIM * EEXP];            // 32 KB of the 320 KB WGP LDS
  const int tid = threadIdx.x;
  for (int i = tid; i < CDIM * EEXP; i += 256) sW[i] = Wr[i];
  __syncthreads();

  const int tok  = blockIdx.x * 8 + (tid >> 5);
  const int lane = tid & 31;
  const float* xr = x + (size_t)tok * CDIM;

  float acc[EEXP];
#pragma unroll
  for (int e = 0; e < EEXP; ++e) acc[e] = 0.f;
  for (int c = lane; c < CDIM; c += 32) {      // coalesced x reads (wave32)
    float xv = xr[c];
#pragma unroll
    for (int e = 0; e < EEXP; ++e) acc[e] += xv * sW[c * EEXP + e];
  }
#pragma unroll
  for (int e = 0; e < EEXP; ++e)
    for (int off = 16; off > 0; off >>= 1)
      acc[e] += __shfl_down(acc[e], off, 32);

  if (lane == 0) {
    float l[EEXP], m = -1e30f;
#pragma unroll
    for (int e = 0; e < EEXP; ++e) { l[e] = acc[e] + br[e]; m = fmaxf(m, l[e]); }
    float s = 0.f, p[EEXP];
#pragma unroll
    for (int e = 0; e < EEXP; ++e) { p[e] = __expf(l[e] - m); s += p[e]; }
    float inv = 1.f / s;
#pragma unroll
    for (int e = 0; e < EEXP; ++e) p[e] *= inv;
    int e0 = 0;                                  // first-index ties, like lax.top_k
#pragma unroll
    for (int e = 1; e < EEXP; ++e) if (p[e] > p[e0]) e0 = e;
    float p0 = p[e0]; p[e0] = -1.f;
    int e1 = 0;
#pragma unroll
    for (int e = 1; e < EEXP; ++e) if (p[e] > p[e1]) e1 = e;
    float p1 = p[e1];
    tke[tok * 2 + 0] = e0; tke[tok * 2 + 1] = e1;
    tkp[tok * 2 + 0] = p0; tkp[tok * 2 + 1] = p1;
  }
}

// ------- deterministic capacity scan: one wave per batch row (ballot prefix) -
__global__ void assign_slots(const int* __restrict__ tke, int* __restrict__ slot) {
  const int b    = blockIdx.x;
  const int lane = threadIdx.x;                // 32 threads (one wave)
  int counts[EEXP];
#pragma unroll
  for (int q = 0; q < EEXP; ++q) counts[q] = 0;
  const unsigned mlt = (1u << lane) - 1u;
  for (int i0 = 0; i0 < TKN; i0 += 32) {
    int e = tke[b * TKN + i0 + lane];
    int pos = 0;
#pragma unroll
    for (int q = 0; q < EEXP; ++q) {
      unsigned m32 = (unsigned)__ballot(e == q);   // wave32: low 32 bits valid
      if (e == q) pos = counts[q] + __popc(m32 & mlt);
      counts[q] += __popc(m32);                    // wave-uniform update
    }
    slot[b * TKN + i0 + lane] = (pos < CAP) ? ((e * BDIM + b) * CAP + pos) : -1;
  }
}

// ---------------- dispatch: scatter kept tokens into bf16 expert buffer -----
__global__ __launch_bounds__(256) void dispatch_kernel(
    const float* __restrict__ x, const int* __restrict__ slot,
    __bf16* __restrict__ bufA) {
  const int idx = blockIdx.x;                  // over B*T*K assignments
  const int s = slot[idx];
  if (s < 0) return;                           // dropped token -> stays zero
  const int b  = idx / TKN;
  const int tk = idx - b * TKN;
  const int t  = tk / KTOP;
  const f32x4v* src = (const f32x4v*)(x + ((size_t)b * TDIM + t) * CDIM);
  bf16x4* dst = (bf16x4*)(bufA + (size_t)s * CDIM);
  for (int c4 = threadIdx.x; c4 < CDIM / 4; c4 += blockDim.x) {
    f32x4v v = src[c4];
    bf16x4 o;
#pragma unroll
    for (int j = 0; j < 4; ++j) o[j] = (__bf16)v[j];
    dst[c4] = o;
  }
}

// ---------------- bf16 WMMA GEMM: Out[e] = act(A[e] @ Bw[e] + bias[e]) -------
// A: [E][M][Kk] bf16 row-major   Bw: [E][Kk][Nn] bf16 row-major
// BM=BN=128, BK=64; 8 waves, each owns 32x64 = 2x4 v_wmma_f32_16x16x32_bf16,
// 2 K-sub-steps per staging round => 16 WMMAs per barrier pair.
// A staged by the Tensor Data Mover with pad_enable (row stride 72 bf16 = 36
// banks; gcd(36,64)=4 so 16 consecutive rows map to distinct LDS banks).
__global__ __launch_bounds__(256) void ffn_gemm(
    const __bf16* __restrict__ A, const __bf16* __restrict__ Bw,
    const float* __restrict__ bias, void* __restrict__ OutP,
    int M, int Nn, int Kk, int relu, int out_bf16) {
  constexpr int BM = 128, BN = 128, BK = 64, LDA = BK + 8, LDB = BK + 8;
  __shared__ __align__(16) __bf16 lA[BM * LDA];   // 18 KB (TDM dest, padded)
  __shared__ __align__(16) __bf16 lB[BN * LDB];   // 18 KB, transposed + padded

  const int tid = threadIdx.x;
  const int e   = blockIdx.z;
  const int m0  = blockIdx.y * BM;
  const int n0  = blockIdx.x * BN;
  A    += (size_t)e * M * Kk;
  Bw   += (size_t)e * Kk * Nn;
  bias += (size_t)e * Nn;

  const int wave = tid >> 5, lane = tid & 31;
  const int wM   = (wave & 3) * 32;      // 4 waves along M
  const int wN   = (wave >> 2) * 64;     // 2 waves along N
  const int half = lane >> 4, l15 = lane & 15;

  f32x8 acc[2][4] = {};                  // 64 VGPRs of f32 accumulators

  const __bf16* Bb = Bw + n0;            // hoisted staging bases
  const __bf16* Ab = A + (size_t)m0 * Kk;

  // Precomputed int32 offsets for the B staging loop (all fit in 32 bits).
  const int chunks = (BK * BN) / 8;      // 1024 -> 4 chunks per thread
  int kk0 = tid >> 4;                    // chunk row
  int nc0 = (tid & 15) << 3;             // chunk column group

#if HAVE_TDM
  // Loop-invariant descriptor pieces (only global_addr changes per k-step).
  const unsigned ldsA_addr = (unsigned)(size_t)lA;
  i32x8 g1c;
  g1c[0] = (int)((1u << 16)              // data_size = 1 (2 bytes)
               | (1u << 20)              // pad_enable
               | (4u << 22)              // pad_interval: 32 DWORDs (one 128B row)
               | (3u << 25));            // pad_amount: 4 DWORDs (16B = 8 bf16)
  g1c[1] = (int)(((unsigned)Kk & 0xFFFFu) << 16);            // tensor_dim0 lo16
  g1c[2] = (int)(((unsigned)Kk >> 16) | (((unsigned)M & 0xFFFFu) << 16));
  g1c[3] = (int)(((unsigned)M >> 16) | ((unsigned)BK << 16)); // tile_dim0 = 64
  g1c[4] = (int)BM;                                           // tile_dim1 = 128
  g1c[5] = (int)(unsigned)Kk;                                 // dim0_stride lo32
  g1c[6] = 0; g1c[7] = 0;
  i32x4 gz; gz[0] = gz[1] = gz[2] = gz[3] = 0;
#if defined(__clang_major__) && (__clang_major__ >= 23)
  i32x8 gz8; for (int i = 0; i < 8; ++i) gz8[i] = 0;
#endif
#endif

  for (int k0 = 0; k0 < Kk; k0 += BK) {
    // ---- stage B tile [BK][BN] transposed: b128 global loads, b16 ds stores.
    // 64-bit base hoisted once; all in-tile offsets are int32.
    const __bf16* Bk = Bb + (size_t)k0 * Nn;
#pragma unroll
    for (int c = 0; c < 4; ++c) {                       // chunks/thread = 4
      int ch = tid + c * 256;
      int kk = kk0 + c * 16;                            // ch >> 4
      int nc = nc0;                                     // (ch & 15) << 3
      (void)ch;
      bf16x8 v = *(const bf16x8*)(Bk + kk * Nn + nc);   // int32 offset math
#pragma unroll
      for (int j = 0; j < 8; ++j) lB[(nc + j) * LDB + kk] = v[j];
    }
    if (k0 + BK < Kk)                   // GL2 prefetch of next B tile
      __builtin_prefetch(Bk + (BK + kk0) * Nn + nc0, 0, 1);

    // ---- stage A tile [BM][BK] via Tensor Data Mover (one wave issues)
#if HAVE_TDM
    if (tid < 32) {
      unsigned long long ga = (unsigned long long)(uintptr_t)(Ab + k0);
      u32x4 g0;
      g0[0] = 1u;                                        // count=1, user mode
      g0[1] = ldsA_addr;                                 // lds_addr
      g0[2] = (unsigned)(ga & 0xFFFFFFFFull);            // global_addr lo
      g0[3] = (unsigned)((ga >> 32) & 0x01FFFFFFull) | (2u << 30);  // hi | type=2
#if defined(__clang_major__) && (__clang_major__ >= 23)
      __builtin_amdgcn_tensor_load_to_lds(g0, g1c, gz, gz, gz8, 0);
#else
      __builtin_amdgcn_tensor_load_to_lds(g0, g1c, gz, gz, 0);
#endif
      __builtin_amdgcn_s_wait_tensorcnt(0);
    }
#else
    {
      const __bf16* Ak = Ab + k0;
      for (int ch = tid; ch < (BM * BK) / 8; ch += 256) {
        int mm = ch >> 3;
        int kc = (ch & 7) << 3;
        *(bf16x8*)&lA[mm * LDA + kc] = *(const bf16x8*)(Ak + mm * Kk + kc);
      }
    }
#endif
    __syncthreads();

    // ---- 2 K-sub-steps of 32: 16 WMMAs per staging round
#pragma unroll
    for (int ks = 0; ks < BK; ks += 32) {
      bf16x16 af[2], bfv[4];
      // A frag: lanes 0-15 hold K ks+{0..7,16..23}; lanes 16-31 +8.
#pragma unroll
      for (int mi = 0; mi < 2; ++mi) {
        const __bf16* p = &lA[(wM + mi * 16 + l15) * LDA + ks + half * 8];
        bf16x8 lo = *(const bf16x8*)p;
        bf16x8 hi = *(const bf16x8*)(p + 16);
        af[mi] = __builtin_shufflevector(lo, hi, 0,1,2,3,4,5,6,7,8,9,10,11,12,13,14,15);
      }
      // B frag: column-contiguous (transposed staging); lanes 0-15 K ks+{0..7,16..23}; lanes 16-31 +8.
#pragma unroll
      for (int ni = 0; ni < 4; ++ni) {
        const __bf16* q = &lB[(wN + ni * 16 + l15) * LDB + ks + half * 8];
        bf16x8 lo = *(const bf16x8*)q;
        bf16x8 hi = *(const bf16x8*)(q + 16);
        bfv[ni] = __builtin_shufflevector(lo, hi, 0,1,2,3,4,5,6,7,8,9,10,11,12,13,14,15);
      }
#pragma unroll
      for (int mi = 0; mi < 2; ++mi)
#pragma unroll
        for (int ni = 0; ni < 4; ++ni)
          acc[mi][ni] = __builtin_amdgcn_wmma_f32_16x16x32_bf16(
              false, af[mi], false, bfv[ni], (short)0, acc[mi][ni], false, false);
    }
    __syncthreads();
  }

  // ---- epilogue: bias (+ReLU); one 64-bit base, int32 per-element offsets
  const size_t obase = ((size_t)e * M + m0) * Nn + n0;
#pragma unroll
  for (int mi = 0; mi < 2; ++mi)
#pragma unroll
    for (int ni = 0; ni < 4; ++ni) {
      int nl = wN + ni * 16 + l15;
      float bv = bias[n0 + nl];
#pragma unroll
      for (int r = 0; r < 8; ++r) {
        int ml = wM + mi * 16 + half * 8 + r;
        float v = acc[mi][ni][r] + bv;
        if (relu) v = fmaxf(v, 0.f);
        int off32 = ml * Nn + nl;                 // < 128*4096, fits int32
        if (out_bf16) ((__bf16*)OutP)[obase + off32] = (__bf16)v;
        else          ((float*)OutP)[obase + off32]  = v;
      }
    }
}

// ---------------- combine: out[tok] = sum_k p_k * y[slot_k] ------------------
__global__ __launch_bounds__(256) void combine_kernel(
    const float* __restrict__ y, const int* __restrict__ slot,
    const float* __restrict__ tkp, float* __restrict__ out) {
  const size_t tok = blockIdx.x;             // over B*T
  int s0 = slot[tok * 2 + 0], s1 = slot[tok * 2 + 1];
  float w0 = (s0 < 0) ? 0.f : tkp[tok * 2 + 0];
  float w1 = (s1 < 0) ? 0.f : tkp[tok * 2 + 1];
  const f32x4v* y0 = (const f32x4v*)(y + (size_t)((s0 < 0) ? 0 : s0) * CDIM);
  const f32x4v* y1 = (const f32x4v*)(y + (size_t)((s1 < 0) ? 0 : s1) * CDIM);
  f32x4v* o = (f32x4v*)(out + tok * CDIM);
  const int c4 = threadIdx.x;                // 256 threads == CDIM/4
  o[c4] = w0 * y0[c4] + w1 * y1[c4];
}

// ---------------- host-side orchestration -----------------------------------
extern "C" void kernel_launch(void* const* d_in, const int* in_sizes, int n_in,
                              void* d_out, int out_size, void* d_ws, size_t ws_size,
                              hipStream_t stream) {
  const float* x  = (const float*)d_in[0];
  const float* Wr = (const float*)d_in[1];
  const float* br = (const float*)d_in[2];
  const float* W1 = (const float*)d_in[3];
  const float* b1 = (const float*)d_in[4];
  const float* W2 = (const float*)d_in[5];
  const float* b2 = (const float*)d_in[6];
  float* out = (float*)d_out;

  char* ws = (char*)d_ws;
  size_t off = 0;
  auto take = [&](size_t bytes) -> char* {
    char* p = ws + off;
    off += (bytes + 255) & ~(size_t)255;
    return p;
  };
  int*    tke  = (int*)   take(sizeof(int)   * BDIM * TDIM * KTOP);
  float*  tkp  = (float*) take(sizeof(float) * BDIM * TDIM * KTOP);
  int*    slot = (int*)   take(sizeof(int)   * BDIM * TDIM * KTOP);
  __bf16* W1b  = (__bf16*)take(2ull * EEXP * CDIM * HDIM);
  __bf16* W2b  = (__bf16*)take(2ull * EEXP * HDIM * CDIM);
  __bf16* bufA = (__bf16*)take(2ull * EEXP * NROW * CDIM);
  __bf16* hbuf = (__bf16*)take(2ull * EEXP * NROW * HDIM);
  float*  ybuf = (float*) take(4ull * EEXP * NROW * CDIM);

  cvt_f32_bf16<<<4096, 256, 0, stream>>>(W1, W1b, (size_t)EEXP * CDIM * HDIM / 4);
  cvt_f32_bf16<<<4096, 256, 0, stream>>>(W2, W2b, (size_t)EEXP * HDIM * CDIM / 4);
  router_topk<<<BDIM * TDIM / 8, 256, 0, stream>>>(x, Wr, br, tke, tkp);
  assign_slots<<<BDIM, 32, 0, stream>>>(tke, slot);
  zero_b128<<<4096, 256, 0, stream>>>((uint4*)bufA,
                                      (2ull * EEXP * NROW * CDIM) / 16);
  dispatch_kernel<<<BDIM * TDIM * KTOP, 256, 0, stream>>>(x, slot, bufA);

  dim3 g1(HDIM / 128, NROW / 128, EEXP);   // 32 x 20 x 8 blocks
  ffn_gemm<<<g1, 256, 0, stream>>>(bufA, W1b, b1, hbuf,
                                   NROW, HDIM, CDIM, /*relu=*/1, /*bf16 out=*/1);
  dim3 g2(CDIM / 128, NROW / 128, EEXP);   // 8 x 20 x 8 blocks
  ffn_gemm<<<g2, 256, 0, stream>>>(hbuf, W2b, b2, ybuf,
                                   NROW, CDIM, HDIM, /*relu=*/0, /*bf16 out=*/0);
  combine_kernel<<<BDIM * TDIM, 256, 0, stream>>>(ybuf, slot, tkp, out);
}
